// SmartMoneyEncoder_60421599920316
// MI455X (gfx1250) — compile-verified, matrix operand order checked by reading, not run on hardware
//
#include <hip/hip_runtime.h>
#include <hip/hip_bf16.h>
#include <math.h>

// ---------------------------------------------------------------------------
// CDNA5 (gfx1250) fused GNN forward: ResGCN -> EdgeGAT -> TGN -> LayerNorm
// All dense GEMMs use v_wmma_f32_16x16x32_bf16 (wave32, 16x16x32 tiles).
// LDS tiles are stored pre-swizzled in WMMA fragment layout so every
// fragment load is one contiguous 32B vector read (2x ds_load_b128).
// Global staging uses float4 (global_load_b128) with hoisted guards.
// Requires (by construction of all call sites): K % 32 == 0, N % 128 == 0.
// ---------------------------------------------------------------------------

typedef __attribute__((ext_vector_type(16))) __bf16 v16bf;
typedef __attribute__((ext_vector_type(8)))  float  v8f;

#define HID 256
#define OUT_DIM 128
#define EDIM 16
#define TDIM 16
#define MSG_K (HID + EDIM + TDIM)   // 288
#define EPS 1e-5f

#define BM 128
#define BN 128
#define BK 32

#define FLAG_ACC  1
#define FLAG_RELU 2
#define FLAG_BIAS 4

#define CDIV(a, b) (((a) + (b) - 1) / (b))

// ---------------------------------------------------------------------------
// WMMA fragment swizzle (CDNA5 §7.12.2, wave32, 16-bit operands)
// A 16x32: lane = m + 16*half holds vgpr v -> K pair kp = (v<4?2v:16+2(v-4))+8*half
// B 32x16: lane = n + 16*half, same K-pair striping.
// Inverse map for K index k in [0,32):
//   half = (k>>3)&1 ; pr = k>>1 ; v = pr<8 ? pr&3 : 4+(pr&3) ; elem = 2v+(k&1)
// LDS layout: frag-major [frag][lane][16 bf16] -> 32B contiguous per lane.
// ---------------------------------------------------------------------------
__device__ __forceinline__ int frag_elem(int k) {
  const int pr = k >> 1;
  const int v = (pr < 8) ? (pr & 3) : (4 + (pr & 3));
  return 2 * v + (k & 1);
}
__device__ __forceinline__ int frag_half(int k) { return (k >> 3) & 1; }

__device__ __forceinline__ v8f zero8() {
  v8f z;
#pragma unroll
  for (int i = 0; i < 8; ++i) z[i] = 0.0f;
  return z;
}

__device__ __forceinline__ void load16(const float* __restrict__ p, float* vals) {
  const float4* p4 = (const float4*)p;
#pragma unroll
  for (int q = 0; q < 4; ++q) {
    float4 t = p4[q];
    vals[4 * q + 0] = t.x;
    vals[4 * q + 1] = t.y;
    vals[4 * q + 2] = t.z;
    vals[4 * q + 3] = t.w;
  }
}

// order-preserving float <-> uint for atomicMax-based segment_max
__device__ __forceinline__ unsigned f2ord(float f) {
  unsigned u = __float_as_uint(f);
  return (u & 0x80000000u) ? ~u : (u | 0x80000000u);
}
__device__ __forceinline__ float ord2f(unsigned u) {
  return (u & 0x80000000u) ? __uint_as_float(u & 0x7FFFFFFFu) : __uint_as_float(~u);
}

// ---------------------------------------------------------------------------
// Generic bf16-WMMA GEMM: C[M,N] (= / +=) A[M,K] @ B[K,N] (+bias) (relu)
// Block = 256 threads (8 waves). Block tile 128x128, wave tile 32x64.
// fp32 -> bf16 conversion + fragment swizzle happen while staging into LDS.
// ---------------------------------------------------------------------------
__global__ __launch_bounds__(256) void wmma_gemm_kernel(
    const float* __restrict__ A, const float* __restrict__ Bm,
    const float* __restrict__ bias, float* __restrict__ C,
    int M, int N, int K, int flags) {
  // [8 row-frags][32 lanes][16 elems] and [8 col-frags][32 lanes][16 elems]
  __shared__ __attribute__((aligned(32))) __bf16 sA[8 * 32 * 16];
  __shared__ __attribute__((aligned(32))) __bf16 sB[8 * 32 * 16];
  const int tid = threadIdx.x;
  const int lane = tid & 31;
  const int wave = tid >> 5;
  const int rowBase = blockIdx.y * BM;
  const int colBase = blockIdx.x * BN;

  v8f acc[2][4];
#pragma unroll
  for (int i = 0; i < 2; ++i)
#pragma unroll
    for (int j = 0; j < 4; ++j) acc[i][j] = zero8();

  for (int kk = 0; kk < K; kk += BK) {
    {  // stage A tile (128x32) swizzled; hoisted row guard, float4 loads
      const int r = tid >> 1, cb = (tid & 1) * 16;
      const int gr = rowBase + r;
      const int ifrag = r >> 4, m = r & 15;
      float vals[16];
      if (gr < M) {
        const float* p = A + (long)gr * K + kk + cb;
        if (kk + BK < K) __builtin_prefetch(p + BK, 0, 1);
        load16(p, vals);
      } else {
#pragma unroll
        for (int q = 0; q < 16; ++q) vals[q] = 0.0f;
      }
#pragma unroll
      for (int c = 0; c < 16; ++c) {
        const int kl = cb + c;
        sA[(ifrag * 32 + (m + 16 * frag_half(kl))) * 16 + frag_elem(kl)] = (__bf16)vals[c];
      }
    }
    {  // stage B tile (32x128) swizzled; K%32==0 and N%128==0 by construction
      const int r = tid >> 3, cb = (tid & 7) * 16;
      const int eidx = frag_elem(r), h = frag_half(r), jf = cb >> 4;
      float vals[16];
      const float* p = Bm + (long)(kk + r) * N + colBase + cb;
      if (kk + BK < K) __builtin_prefetch(p + (long)BK * N, 0, 1);
      load16(p, vals);
#pragma unroll
      for (int c = 0; c < 16; ++c)
        sB[(jf * 32 + (c + 16 * h)) * 16 + eidx] = (__bf16)vals[c];
    }
    __syncthreads();
    const v16bf* sAv = (const v16bf*)sA;
    const v16bf* sBv = (const v16bf*)sB;
    v16bf af[2], bf[4];
#pragma unroll
    for (int i = 0; i < 2; ++i) af[i] = sAv[((wave & 3) * 2 + i) * 32 + lane];
#pragma unroll
    for (int j = 0; j < 4; ++j) bf[j] = sBv[((wave >> 2) * 4 + j) * 32 + lane];
#pragma unroll
    for (int i = 0; i < 2; ++i)
#pragma unroll
      for (int j = 0; j < 4; ++j)
        acc[i][j] = __builtin_amdgcn_wmma_f32_16x16x32_bf16(
            false, af[i], false, bf[j], (short)0, acc[i][j], false, false);
    __syncthreads();
  }

  const int half16 = lane >> 4, n15 = lane & 15;
  const int wm = (wave & 3) * 32, wn = (wave >> 2) * 64;
#pragma unroll
  for (int i = 0; i < 2; ++i) {
#pragma unroll
    for (int j = 0; j < 4; ++j) {
      const int col = colBase + wn + j * 16 + n15;
#pragma unroll
      for (int r = 0; r < 8; ++r) {
        const int row = rowBase + wm + i * 16 + r + half16 * 8;
        if (row < M) {
          float v = acc[i][j][r];
          if (flags & FLAG_BIAS) v += bias[col];
          if (flags & FLAG_ACC) v += C[(long)row * N + col];
          if (flags & FLAG_RELU) v = fmaxf(v, 0.0f);
          C[(long)row * N + col] = v;
        }
      }
    }
  }
}

// ---------------------------------------------------------------------------
// TGN message GEMM, fully fused:
//   A row e = [ hm[src[e]] (256) | edge_attr[e] (16) | cos(t*w+b) (16) ]
//   msg = relu(A @ Wmsg + bmsg);  agg[dst[e]] += msg * invdeg0[dst[e]]
// msg (E x 256, 328 MB > 192 MB L2) is never materialized.
// Since kk steps by 32 and segment boundaries (256,272,288) are 16-aligned,
// each 16-element staging chunk lies entirely inside one segment.
// ---------------------------------------------------------------------------
__global__ __launch_bounds__(256) void wmma_msg_kernel(
    const float* __restrict__ hm, const float* __restrict__ ea,
    const int* __restrict__ ts, const float* __restrict__ tw,
    const float* __restrict__ tb, const float* __restrict__ W,
    const float* __restrict__ bmsg, const int* __restrict__ src,
    const int* __restrict__ dst, const float* __restrict__ invd0,
    float* __restrict__ agg, int E) {
  __shared__ __attribute__((aligned(32))) __bf16 sA[8 * 32 * 16];
  __shared__ __attribute__((aligned(32))) __bf16 sB[8 * 32 * 16];
  const int tid = threadIdx.x;
  const int lane = tid & 31;
  const int wave = tid >> 5;
  const int rowBase = blockIdx.y * BM;
  const int colBase = blockIdx.x * BN;

  v8f acc[2][4];
#pragma unroll
  for (int i = 0; i < 2; ++i)
#pragma unroll
    for (int j = 0; j < 4; ++j) acc[i][j] = zero8();

  for (int kk = 0; kk < MSG_K; kk += BK) {
    {  // stage gathered A tile (segment-hoisted), swizzled
      const int r = tid >> 1, cb = (tid & 1) * 16;
      const int e = rowBase + r;
      const int ifrag = r >> 4, m = r & 15;
      const int kbase = kk + cb;  // 16-aligned chunk start
      float vals[16];
      if (e < E) {
        if (kbase < HID) {  // gathered node features
          const int s = src[e];
          const float* p = hm + (long)s * HID + kbase;
          if (kbase + BK < HID) __builtin_prefetch(p + BK, 0, 1);
          load16(p, vals);
        } else if (kbase < HID + EDIM) {  // edge features (full 16-wide chunk)
          load16(ea + (long)e * EDIM, vals);
        } else {  // cosine time encoding (full 16-wide chunk)
          const float tse = (float)ts[e];
#pragma unroll
          for (int q = 0; q < TDIM; ++q) vals[q] = cosf(tse * tw[q] + tb[q]);
        }
      } else {
#pragma unroll
        for (int q = 0; q < 16; ++q) vals[q] = 0.0f;
      }
#pragma unroll
      for (int c = 0; c < 16; ++c) {
        const int kl = cb + c;
        sA[(ifrag * 32 + (m + 16 * frag_half(kl))) * 16 + frag_elem(kl)] = (__bf16)vals[c];
      }
    }
    {  // stage Wmsg tile (288x256 row-major), swizzled
      const int r = tid >> 3, cb = (tid & 7) * 16;
      const int eidx = frag_elem(r), h = frag_half(r), jf = cb >> 4;
      float vals[16];
      const float* p = W + (long)(kk + r) * HID + colBase + cb;
      if (kk + BK < MSG_K) __builtin_prefetch(p + (long)BK * HID, 0, 1);
      load16(p, vals);
#pragma unroll
      for (int c = 0; c < 16; ++c)
        sB[(jf * 32 + (c + 16 * h)) * 16 + eidx] = (__bf16)vals[c];
    }
    __syncthreads();
    const v16bf* sAv = (const v16bf*)sA;
    const v16bf* sBv = (const v16bf*)sB;
    v16bf af[2], bf[4];
#pragma unroll
    for (int i = 0; i < 2; ++i) af[i] = sAv[((wave & 3) * 2 + i) * 32 + lane];
#pragma unroll
    for (int j = 0; j < 4; ++j) bf[j] = sBv[((wave >> 2) * 4 + j) * 32 + lane];
#pragma unroll
    for (int i = 0; i < 2; ++i)
#pragma unroll
      for (int j = 0; j < 4; ++j)
        acc[i][j] = __builtin_amdgcn_wmma_f32_16x16x32_bf16(
            false, af[i], false, bf[j], (short)0, acc[i][j], false, false);
    __syncthreads();
  }

  const int half16 = lane >> 4, n15 = lane & 15;
  const int wm = (wave & 3) * 32, wn = (wave >> 2) * 64;
#pragma unroll
  for (int i = 0; i < 2; ++i) {
#pragma unroll
    for (int j = 0; j < 4; ++j) {
      const int col = colBase + wn + j * 16 + n15;
#pragma unroll
      for (int r = 0; r < 8; ++r) {
        const int e = rowBase + wm + i * 16 + r + half16 * 8;
        if (e < E) {
          const int d = dst[e];
          const float v = fmaxf(acc[i][j][r] + bmsg[col], 0.0f);
          atomicAdd(&agg[(long)d * HID + col], v * invd0[d]);
        }
      }
    }
  }
}

// ---------------------------------------------------------------------------
// Elementwise / scatter helpers
// ---------------------------------------------------------------------------
__global__ void fill_f32(float* p, float v, long n) {
  long i = (long)blockIdx.x * 256 + threadIdx.x;
  if (i < n) p[i] = v;
}
__global__ void fill_u32(unsigned* p, unsigned v, long n) {
  long i = (long)blockIdx.x * 256 + threadIdx.x;
  if (i < n) p[i] = v;
}

__global__ void deg_count(const int* __restrict__ dst, float* __restrict__ cnt, int E) {
  int e = blockIdx.x * 256 + threadIdx.x;
  if (e < E) atomicAdd(&cnt[dst[e]], 1.0f);
}
// deg (with self-loop) = cnt+1 ; dinv = rsqrt(deg) ; invdeg0 = 1/max(cnt,1)
__global__ void deg_finish(const float* __restrict__ cnt, float* __restrict__ dinv,
                           float* __restrict__ invd0, int N) {
  int i = blockIdx.x * 256 + threadIdx.x;
  if (i < N) {
    float c = cnt[i];
    dinv[i] = rsqrtf(c + 1.0f);
    invd0[i] = 1.0f / fmaxf(c, 1.0f);
  }
}

// GCN sym-normalized scatter: agg[dst] += xw[src]*dinv[src]*dinv[dst] (incl self-loops)
__global__ void gcn_scatter(const float* __restrict__ xw, const int* __restrict__ src,
                            const int* __restrict__ dst, const float* __restrict__ dinv,
                            float* __restrict__ agg, int E, int N) {
  long idx = (long)blockIdx.x * 256 + threadIdx.x;
  long total = (long)(E + N) * (HID / 4);
  if (idx >= total) return;
  int e = (int)(idx >> 6);
  int c = (int)(idx & 63) * 4;
  int s, d;
  if (e < E) { s = src[e]; d = dst[e]; } else { s = d = e - E; }
  float w = dinv[s] * dinv[d];
  const float* row = xw + (long)s * HID + c;
  float* o = agg + (long)d * HID + c;
  atomicAdd(o + 0, row[0] * w);
  atomicAdd(o + 1, row[1] * w);
  atomicAdd(o + 2, row[2] * w);
  atomicAdd(o + 3, row[3] * w);
}

// BN(eval) + ReLU + optional residual
__global__ void bn_relu(const float* __restrict__ agg, const float* __restrict__ b,
                        const float* __restrict__ g, const float* __restrict__ bt,
                        const float* __restrict__ mu, const float* __restrict__ var,
                        const float* __restrict__ res, float* __restrict__ out, int N) {
  long idx = (long)blockIdx.x * 256 + threadIdx.x;
  if (idx >= (long)N * HID) return;
  int c = (int)(idx & (HID - 1));
  float v = agg[idx] + b[c];
  v = (v - mu[c]) * rsqrtf(var[c] + EPS) * g[c] + bt[c];
  v = fmaxf(v, 0.0f);
  if (res) v += res[idx];
  out[idx] = v;
}

// fold GAT edge projection: qe[h*16+k] = sum_c gat_we[k, h*128+c] * aedge[h,c]
__global__ void gat_qe(const float* __restrict__ we, const float* __restrict__ aedge,
                       float* __restrict__ qe) {
  int t = blockIdx.x * 32 + threadIdx.x;
  if (t >= 32) return;
  int h = t >> 4, k = t & 15;
  float s = 0.0f;
  for (int c = 0; c < 128; ++c) s += we[k * HID + h * 128 + c] * aedge[h * 128 + c];
  qe[h * 16 + k] = s;
}

// per-node attention scalars: s_src[n,h] = xw[n,h,:]·asrc[h], same for dst
__global__ void gat_scores(const float* __restrict__ xw, const float* __restrict__ asrc,
                           const float* __restrict__ adst, float* __restrict__ ssrc,
                           float* __restrict__ sdst, int N) {
  int idx = blockIdx.x * 256 + threadIdx.x;
  if (idx >= N * 2) return;
  int n = idx >> 1, h = idx & 1;
  float a = 0.0f, b = 0.0f;
  const float* row = xw + (long)n * HID + h * 128;
  for (int c = 0; c < 128; ++c) {
    float v = row[c];
    a += v * asrc[h * 128 + c];
    b += v * adst[h * 128 + c];
  }
  ssrc[idx] = a;
  sdst[idx] = b;
}

__global__ void gat_logit(const float* __restrict__ ea, const float* __restrict__ qe,
                          const int* __restrict__ src, const int* __restrict__ dst,
                          const float* __restrict__ ssrc, const float* __restrict__ sdst,
                          float* __restrict__ logit, unsigned* __restrict__ mxu,
                          int E, int N) {
  int e = blockIdx.x * 256 + threadIdx.x;
  if (e >= E + N) return;
  int s, d;
  float et0 = 0.0f, et1 = 0.0f;
  if (e < E) {
    s = src[e]; d = dst[e];
    for (int k = 0; k < EDIM; ++k) {
      float a = ea[(long)e * EDIM + k];
      et0 += a * qe[k];
      et1 += a * qe[16 + k];
    }
  } else {
    s = d = e - E;  // self-loop, zero edge feature
  }
#pragma unroll
  for (int h = 0; h < 2; ++h) {
    float l = ssrc[s * 2 + h] + sdst[d * 2 + h] + (h ? et1 : et0);
    l = (l > 0.0f) ? l : 0.2f * l;  // leaky_relu(0.2)
    logit[(long)e * 2 + h] = l;
    atomicMax(&mxu[d * 2 + h], f2ord(l));
  }
}

__global__ void gat_expsum(float* __restrict__ logit, const unsigned* __restrict__ mxu,
                           const int* __restrict__ dst, float* __restrict__ ssum,
                           int E, int N) {
  int e = blockIdx.x * 256 + threadIdx.x;
  if (e >= E + N) return;
  int d = (e < E) ? dst[e] : (e - E);
#pragma unroll
  for (int h = 0; h < 2; ++h) {
    float p = expf(logit[(long)e * 2 + h] - ord2f(mxu[d * 2 + h]));
    logit[(long)e * 2 + h] = p;  // overwrite in place
    atomicAdd(&ssum[d * 2 + h], p);
  }
}

__global__ void gat_scatter(const float* __restrict__ xw, const float* __restrict__ p,
                            const float* __restrict__ ssum, const int* __restrict__ src,
                            const int* __restrict__ dst, float* __restrict__ out,
                            int E, int N) {
  long idx = (long)blockIdx.x * 256 + threadIdx.x;
  long total = (long)(E + N) * (HID / 4);
  if (idx >= total) return;
  int e = (int)(idx >> 6);
  int c = (int)(idx & 63) * 4;
  int h = c >> 7;
  int s, d;
  if (e < E) { s = src[e]; d = dst[e]; } else { s = d = e - E; }
  float alpha = p[(long)e * 2 + h] / (ssum[d * 2 + h] + 1e-16f);
  const float* row = xw + (long)s * HID + c;
  float* o = out + (long)d * HID + c;
  atomicAdd(o + 0, row[0] * alpha);
  atomicAdd(o + 1, row[1] * alpha);
  atomicAdd(o + 2, row[2] * alpha);
  atomicAdd(o + 3, row[3] * alpha);
}

__global__ void add_bias(float* __restrict__ h, const float* __restrict__ bias, long total) {
  long idx = (long)blockIdx.x * 256 + threadIdx.x;
  if (idx < total) h[idx] += bias[(int)(idx & (HID - 1))];
}

// final LayerNorm over 128 cols: one wave32 per row, 4 cols/lane, shfl reduce
__global__ __launch_bounds__(256) void layernorm_out(
    const float* __restrict__ raw, const float* __restrict__ g,
    const float* __restrict__ b, float* __restrict__ out, int N) {
  const int wave = threadIdx.x >> 5, lane = threadIdx.x & 31;
  const int row = blockIdx.x * 8 + wave;
  if (row >= N) return;
  float v[4], s = 0.0f;
#pragma unroll
  for (int j = 0; j < 4; ++j) {
    v[j] = raw[(long)row * OUT_DIM + lane + j * 32];
    s += v[j];
  }
#pragma unroll
  for (int off = 16; off > 0; off >>= 1) s += __shfl_xor(s, off);
  const float mu = s * (1.0f / OUT_DIM);
  float q = 0.0f;
#pragma unroll
  for (int j = 0; j < 4; ++j) {
    float d = v[j] - mu;
    q += d * d;
  }
#pragma unroll
  for (int off = 16; off > 0; off >>= 1) q += __shfl_xor(q, off);
  const float rs = rsqrtf(q * (1.0f / OUT_DIM) + EPS);
#pragma unroll
  for (int j = 0; j < 4; ++j) {
    const int c = lane + j * 32;
    out[(long)row * OUT_DIM + c] = (v[j] - mu) * rs * g[c] + b[c];
  }
}

// ---------------------------------------------------------------------------
// Host-side orchestration
// ---------------------------------------------------------------------------
static inline void gemm(const float* A, const float* B, const float* bias, float* C,
                        int M, int N, int K, int flags, hipStream_t stream) {
  dim3 grid(CDIV(N, BN), CDIV(M, BM));
  wmma_gemm_kernel<<<grid, 256, 0, stream>>>(A, B, bias, C, M, N, K, flags);
}

extern "C" void kernel_launch(void* const* d_in, const int* in_sizes, int n_in,
                              void* d_out, int out_size, void* d_ws, size_t ws_size,
                              hipStream_t stream) {
  const float* x        = (const float*)d_in[0];
  const int*   ei       = (const int*)d_in[1];
  const float* ea       = (const float*)d_in[2];
  const int*   ts       = (const int*)d_in[3];
  const float* memry    = (const float*)d_in[4];
  const float* gcn_w0   = (const float*)d_in[5];
  const float* gcn_b0   = (const float*)d_in[6];
  const float* gcn_w    = (const float*)d_in[7];
  const float* gcn_b    = (const float*)d_in[8];
  const float* bn_gamma = (const float*)d_in[9];
  const float* bn_beta  = (const float*)d_in[10];
  const float* bn_mean  = (const float*)d_in[11];
  const float* bn_var   = (const float*)d_in[12];
  const float* gat_w    = (const float*)d_in[13];
  const float* gat_we   = (const float*)d_in[14];
  const float* gat_asrc = (const float*)d_in[15];
  const float* gat_adst = (const float*)d_in[16];
  const float* gat_aedg = (const float*)d_in[17];
  const float* gat_bias = (const float*)d_in[18];
  const float* time_w   = (const float*)d_in[19];
  const float* time_b   = (const float*)d_in[20];
  const float* tgn_wx   = (const float*)d_in[21];
  const float* tgn_wm   = (const float*)d_in[22];
  const float* tgn_bh   = (const float*)d_in[23];
  const float* tgn_wmsg = (const float*)d_in[24];
  const float* tgn_bmsg = (const float*)d_in[25];
  const float* tgn_wself= (const float*)d_in[26];
  const float* tgn_wagg = (const float*)d_in[27];
  const float* tgn_bupd = (const float*)d_in[28];
  const float* tgn_wout = (const float*)d_in[29];
  const float* tgn_bout = (const float*)d_in[30];
  const float* ln_gamma = (const float*)d_in[31];
  const float* ln_beta  = (const float*)d_in[32];

  const int IN = 128;
  const int MDIM = 128;
  const int N = in_sizes[0] / IN;   // 20000
  const int E = in_sizes[3];        // 320000
  const int* src = ei;
  const int* dst = ei + E;

  // workspace layout (floats)
  float* f = (float*)d_ws;
  float*    cnt   = f;                 // N
  float*    dinv  = f + (size_t)N;     // N
  float*    invd0 = f + 2 * (size_t)N; // N
  float*    ssrc  = f + 3 * (size_t)N; // 2N
  float*    sdst  = f + 5 * (size_t)N; // 2N
  float*    ssum  = f + 7 * (size_t)N; // 2N
  unsigned* mxu   = (unsigned*)(f + 9 * (size_t)N); // 2N
  float*    qe    = f + 11 * (size_t)N;             // 32
  float*    logit = qe + 32;                        // 2*(E+N)
  size_t base2 = 11 * (size_t)N + 32 + 2 * (size_t)(E + N);
  float* bufA = f + base2;                          // N*HID
  float* bufB = bufA + (size_t)N * HID;             // N*HID
  float* bufC = bufB + (size_t)N * HID;             // N*HID

  const long NH = (long)N * HID;
  const long EN64 = (long)(E + N) * (HID / 4);

  // ---- degrees / normalization ----
  fill_f32<<<CDIV(N, 256), 256, 0, stream>>>(cnt, 0.0f, N);
  deg_count<<<CDIV(E, 256), 256, 0, stream>>>(dst, cnt, E);
  deg_finish<<<CDIV(N, 256), 256, 0, stream>>>(cnt, dinv, invd0, N);

  // ---- ResGCN (3 layers, BN eval mode, residual from layer 1) ----
  for (int l = 0; l < 3; ++l) {
    const float* hin = (l == 0) ? x : bufA;
    const int K = (l == 0) ? IN : HID;
    const float* W = (l == 0) ? gcn_w0 : (gcn_w + (size_t)(l - 1) * HID * HID);
    const float* b = (l == 0) ? gcn_b0 : (gcn_b + (size_t)(l - 1) * HID);
    gemm(hin, W, nullptr, bufB, N, HID, K, 0, stream);                     // xw
    fill_f32<<<CDIV(NH, 256), 256, 0, stream>>>(bufC, 0.0f, NH);
    gcn_scatter<<<CDIV(EN64, 256), 256, 0, stream>>>(bufB, src, dst, dinv, bufC, E, N);
    bn_relu<<<CDIV(NH, 256), 256, 0, stream>>>(
        bufC, b, bn_gamma + (size_t)l * HID, bn_beta + (size_t)l * HID,
        bn_mean + (size_t)l * HID, bn_var + (size_t)l * HID,
        (l == 0) ? nullptr : bufA, bufA, N);
  }

  // ---- EdgeGAT (2 heads, edge projection algebraically folded to qe[2,16]) ----
  gemm(bufA, gat_w, nullptr, bufB, N, HID, HID, 0, stream);                // xw
  gat_qe<<<1, 32, 0, stream>>>(gat_we, gat_aedg, qe);
  gat_scores<<<CDIV(N * 2, 256), 256, 0, stream>>>(bufB, gat_asrc, gat_adst, ssrc, sdst, N);
  fill_u32<<<CDIV(2 * N, 256), 256, 0, stream>>>(mxu, 0x007FFFFFu /* ord(-inf) */, 2 * N);
  fill_f32<<<CDIV(2 * N, 256), 256, 0, stream>>>(ssum, 0.0f, 2 * N);
  gat_logit<<<CDIV(E + N, 256), 256, 0, stream>>>(ea, qe, src, dst, ssrc, sdst, logit, mxu, E, N);
  gat_expsum<<<CDIV(E + N, 256), 256, 0, stream>>>(logit, mxu, dst, ssum, E, N);
  fill_f32<<<CDIV(NH, 256), 256, 0, stream>>>(bufC, 0.0f, NH);
  gat_scatter<<<CDIV(EN64, 256), 256, 0, stream>>>(bufB, logit, ssum, src, dst, bufC, E, N);
  add_bias<<<CDIV(NH, 256), 256, 0, stream>>>(bufC, gat_bias, NH);         // h = bufC

  // ---- TGN memory read ----
  gemm(bufC, tgn_wx, nullptr, bufB, N, HID, HID, 0, stream);
  gemm(memry, tgn_wm, tgn_bh, bufB, N, HID, MDIM, FLAG_ACC | FLAG_BIAS | FLAG_RELU, stream);
  // hm = bufB

  // ---- TGN message passing (2 layers; msg GEMM fused with gather+scatter) ----
  for (int l = 0; l < 2; ++l) {
    float* hm_cur  = (l == 0) ? bufB : bufC;
    float* hm_next = (l == 0) ? bufC : bufB;
    fill_f32<<<CDIV(NH, 256), 256, 0, stream>>>(bufA, 0.0f, NH);           // agg
    dim3 mgrid(CDIV(HID, BN), CDIV(E, BM));
    wmma_msg_kernel<<<mgrid, 256, 0, stream>>>(
        hm_cur, ea, ts, time_w, time_b,
        tgn_wmsg + (size_t)l * MSG_K * HID, tgn_bmsg + (size_t)l * HID,
        src, dst, invd0, bufA, E);
    gemm(hm_cur, tgn_wself + (size_t)l * HID * HID, nullptr, hm_next, N, HID, HID, 0, stream);
    gemm(bufA, tgn_wagg + (size_t)l * HID * HID, tgn_bupd + (size_t)l * HID,
         hm_next, N, HID, HID, FLAG_ACC | FLAG_BIAS | FLAG_RELU, stream);
  }
  // final hm = bufB

  // ---- output projection + LayerNorm ----
  gemm(bufB, tgn_wout, tgn_bout, bufA, N, OUT_DIM, HID, FLAG_BIAS, stream);
  layernorm_out<<<CDIV(N, 8), 256, 0, stream>>>(bufA, ln_gamma, ln_beta, (float*)d_out, N);
}
